// ExpertGraphConv_7060926234636
// MI455X (gfx1250) — matrix operand order repaired and checked by reading, not compiled, not run on hardware
//
#include <hip/hip_runtime.h>
#include <hip/hip_bf16.h>
#include <math.h>

// ---------------------------------------------------------------------------
// ExpertGraphConv fused kernel for gfx1250 (MI455X).
//   out = gelu( (strength @ x) @ Wn^T + x @ Ws^T + bn + bs )
// Single K=1024 bf16 WMMA GEMM: A = [msg | x], B = [Wn ; Ws]^T (prepacked).
// Streaming tensors (x, out) use non-temporal hints to keep B resident in L2.
// ---------------------------------------------------------------------------

typedef __attribute__((ext_vector_type(4)))  float  v4f;
typedef __attribute__((ext_vector_type(8)))  float  v8f;
typedef __attribute__((ext_vector_type(8)))  __bf16 v8bf;
typedef __attribute__((ext_vector_type(16))) __bf16 v16bf;

#define D_DIM 512
#define TOKS  8
#define ROWS  64            // TOKS * E(=8) rows per block
#define LDK   520           // padded bf16 row pitch in LDS (bank-conflict free)
#define KT_N  32            // K tiles (K = 1024, 32 per WMMA)
#define NT_N  32            // N tiles (N = 512, 16 per WMMA)

__device__ __forceinline__ unsigned short f2bf(float f) {
  unsigned u = __float_as_uint(f);
  u += 0x7FFFu + ((u >> 16) & 1u);          // round-to-nearest-even
  return (unsigned short)(u >> 16);
}
__device__ __forceinline__ float bf2f(unsigned short h) {
  return __uint_as_float(((unsigned)h) << 16);
}
__device__ __forceinline__ float sigmoidf_(float x) { return 1.0f / (1.0f + __expf(-x)); }
__device__ __forceinline__ float gelu_exact(float x) {
  return 0.5f * x * (1.0f + erff(x * 0.70710678118654752f));
}

// ---------------------------------------------------------------------------
// Prep: pack Wcat = [Wn ; Ws] (as B = Wcat^T, 1024x512) into bf16 fragments
// in this kernel's per-lane v16bf B-operand order (contiguous k, as A uses):
//   idx = ((nt*32 + kt)*32 + lane)*16 + h,  k = kt*32 + (lane>>4)*16 + h,
//   n = nt*16 + (lane&15)
// ---------------------------------------------------------------------------
__global__ __launch_bounds__(256) void prep_wcat_bf16(
    const float* __restrict__ Wn, const float* __restrict__ Ws,
    unsigned short* __restrict__ Bfrag) {
  int idx  = blockIdx.x * 256 + threadIdx.x;      // < 32*32*32*16 = 524288
  int h    = idx & 15;
  int lane = (idx >> 4) & 31;
  int kt   = (idx >> 9) & 31;
  int nt   = idx >> 14;
  int k    = kt * 32 + (lane >> 4) * 16 + h;
  int n    = nt * 16 + (lane & 15);
  float w  = (k < 512) ? Wn[n * 512 + k] : Ws[n * 512 + (k - 512)];
  Bfrag[idx] = f2bf(w);
}

// ---------------------------------------------------------------------------
// Fused main kernel: 1 block = 8 tokens = 64 rows. 256 threads = 8 wave32.
// ---------------------------------------------------------------------------
__global__ __launch_bounds__(256) void expert_graph_conv_kernel(
    const float* __restrict__ x,
    const float* __restrict__ b_neighbor,
    const float* __restrict__ b_self,
    const float* __restrict__ w_msg,
    const float* __restrict__ b_msg,
    const float* __restrict__ adj_logits,
    const unsigned short* __restrict__ Bfrag,
    float* __restrict__ out) {
  extern __shared__ char smem[];
  unsigned short* Xs = (unsigned short*)smem;                      // ROWS*LDK bf16
  unsigned short* Ms = (unsigned short*)(smem + ROWS * LDK * 2);   // ROWS*LDK bf16
  float* WAs = (float*)(smem + 2 * ROWS * LDK * 2);                // 512
  float* WBs = WAs + 512;                                          // 512
  float* As  = WBs + 512;                                          // 64
  float* Bs  = As + ROWS;                                          // 64
  float* Ss  = Bs + ROWS;                                          // 8*64 strengths
  float* Pa  = Ss + TOKS * 64;                                     // 256 partials
  float* Pb  = Pa + 256;                                           // 256 partials

  const int tid = threadIdx.x;
  const size_t base = (size_t)blockIdx.x * ROWS * D_DIM;

  // Stage 0: w_msg halves into LDS
  for (int d = tid; d < 512; d += 256) { WAs[d] = w_msg[d]; WBs[d] = w_msg[512 + d]; }

  // Stage 1: coalesced NT float4 load of x chunk -> bf16 LDS (padded pitch).
  // x is streamed exactly once: non-temporal keeps the packed weights in L2.
  const v4f* xg = (const v4f*)(x + base);
  for (int i = tid; i < ROWS * D_DIM / 4; i += 256) {
    v4f v = __builtin_nontemporal_load(xg + i);
    int e0 = i << 2;
    int r  = e0 >> 9;         // / 512
    int d  = e0 & 511;
    unsigned a0 = (unsigned)f2bf(v.x) | ((unsigned)f2bf(v.y) << 16);
    unsigned a1 = (unsigned)f2bf(v.z) | ((unsigned)f2bf(v.w) << 16);
    *(uint2*)(Xs + r * LDK + d) = make_uint2(a0, a1);
  }
  __syncthreads();

  // Stage 2a: partial row dots, 4 threads per row (balanced across the block)
  {
    int r = tid >> 2, q = tid & 3;
    const unsigned short* xr = Xs + r * LDK + q * 128;
    const float* wa = WAs + q * 128;
    const float* wb = WBs + q * 128;
    float sa = 0.f, sb = 0.f;
    for (int d = 0; d < 128; ++d) {
      float xv = bf2f(xr[d]);
      sa += xv * wa[d];
      sb += xv * wb[d];
    }
    Pa[tid] = sa; Pb[tid] = sb;
  }
  __syncthreads();
  if (tid < ROWS) {
    int b4 = tid * 4;
    As[tid] = (Pa[b4] + Pa[b4 + 1]) + (Pa[b4 + 2] + Pa[b4 + 3]);
    Bs[tid] = (Pb[b4] + Pb[b4 + 1]) + (Pb[b4 + 2] + Pb[b4 + 3]);
  }
  __syncthreads();

  // Stage 3: strength[tok][i][j] = sigmoid(adj)*sigmoid(a_i+b_j+bm) * (i!=j)
  float bm = b_msg[0];
  for (int idx = tid; idx < TOKS * 64; idx += 256) {
    int tok = idx >> 6, ij = idx & 63, i = ij >> 3, j = ij & 7;
    float s = 0.0f;
    if (i != j) {
      float adjv = sigmoidf_(adj_logits[ij]);
      float cont = sigmoidf_(As[tok * 8 + i] + Bs[tok * 8 + j] + bm);
      s = adjv * cont;
    }
    Ss[idx] = s;
  }
  __syncthreads();

  // Stage 4: msg[r][d] = sum_j strength * x  (K=8, <1% of FLOPs -> VALU)
  for (int idx = tid; idx < ROWS * D_DIM; idx += 256) {
    int r = idx >> 9, d = idx & 511;
    int tok = r >> 3;
    const float* sr = Ss + tok * 64 + (r & 7) * 8;
    const unsigned short* xb = Xs + tok * 8 * LDK + d;
    float acc = 0.f;
#pragma unroll
    for (int j = 0; j < 8; ++j) acc += sr[j] * bf2f(xb[j * LDK]);
    Ms[r * LDK + d] = f2bf(acc);
  }
  __syncthreads();

  // Stage 5: WMMA GEMM. M=64 (4 tiles), N=512 (32 tiles), K=1024 (32 steps).
  // Wave w: mtile = w&3, owns 16 N-tiles processed TWO at a time (two
  // independent accumulator chains sharing each A fragment) with a one-step
  // B prefetch rotation so global loads overlap the WMMAs.
  const int lane  = tid & 31;
  const int wv    = tid >> 5;
  const int mt    = wv & 3;
  const int lrow  = lane & 15;
  const int lhalf = lane >> 4;
  const int m     = mt * 16 + lrow;
  const unsigned short* Amsg = Ms + m * LDK;
  const unsigned short* Ax   = Xs + m * LDK;
  const int ntBase = (wv >> 2) * 16;

#pragma unroll 1
  for (int p = 0; p < 8; ++p) {
    const int nt0 = ntBase + 2 * p;
    const unsigned short* bp0 = Bfrag + (size_t)nt0 * (KT_N * 512) + lane * 16;
    const unsigned short* bp1 = bp0 + KT_N * 512;
    v8f c0 = {}, c1 = {};
    v16bf b0 = *(const v16bf*)bp0;
    v16bf b1 = *(const v16bf*)bp1;
#pragma unroll
    for (int kt = 0; kt < KT_N; ++kt) {
      // A fragment: lane row = lane&15; lanes>=16 get K+16;
      // h -> k+h (contiguous, matches packed B)  => two contiguous b128 LDS loads
      const unsigned short* arow = (kt < 16) ? Amsg : Ax;   // K<512: msg, else x
      int kloc = (kt & 15) * 32 + lhalf * 16;
      v8bf lo = *(const v8bf*)(arow + kloc);
      v8bf hi = *(const v8bf*)(arow + kloc + 8);
      v16bf a;
#pragma unroll
      for (int t = 0; t < 8; ++t) { a[t] = lo[t]; a[t + 8] = hi[t]; }
      v16bf nb0 = b0, nb1 = b1;
      if (kt + 1 < KT_N) {                 // prefetch next K-step's B operands
        nb0 = *(const v16bf*)(bp0 + (kt + 1) * 512);
        nb1 = *(const v16bf*)(bp1 + (kt + 1) * 512);
      }
      c0 = __builtin_amdgcn_wmma_f32_16x16x32_bf16(
          false, a, false, b0, (short)0, c0, false, false);
      c1 = __builtin_amdgcn_wmma_f32_16x16x32_bf16(
          false, a, false, b1, (short)0, c1, false, false);
      b0 = nb0; b1 = nb1;
    }
    // Epilogue: C layout VGPR rr, lanes>=16 -> M+8; biases + exact GELU.
    // Output is written exactly once: non-temporal stores spare L2.
    int n0 = nt0 * 16 + lrow;
    float bias0 = b_neighbor[n0] + b_self[n0];
    float bias1 = b_neighbor[n0 + 16] + b_self[n0 + 16];
#pragma unroll
    for (int rr = 0; rr < 8; ++rr) {
      int mr = mt * 16 + lhalf * 8 + rr;
      float v0 = gelu_exact(c0[rr] + bias0);
      float v1 = gelu_exact(c1[rr] + bias1);
      __builtin_nontemporal_store(v0, &out[base + (size_t)mr * D_DIM + n0]);
      __builtin_nontemporal_store(v1, &out[base + (size_t)mr * D_DIM + n0 + 16]);
    }
  }
}

extern "C" void kernel_launch(void* const* d_in, const int* in_sizes, int n_in,
                              void* d_out, int out_size, void* d_ws, size_t ws_size,
                              hipStream_t stream) {
  const float* x   = (const float*)d_in[0];   // expert_features (8,2048,8,512)
  const float* Wn  = (const float*)d_in[1];   // W_neighbor (512,512)
  const float* bn  = (const float*)d_in[2];   // b_neighbor (512)
  const float* Ws  = (const float*)d_in[3];   // W_self (512,512)
  const float* bs  = (const float*)d_in[4];   // b_self (512)
  const float* wm  = (const float*)d_in[5];   // w_msg (1024)
  const float* bm  = (const float*)d_in[6];   // b_msg (1)
  const float* al  = (const float*)d_in[7];   // adjacency_logits (8,8)
  unsigned short* Bfrag = (unsigned short*)d_ws;   // 1 MB packed bf16 B operand
  float* out = (float*)d_out;

  // Per-call weight packing (deterministic, graph-capture safe).
  prep_wcat_bf16<<<524288 / 256, 256, 0, stream>>>(Wn, Ws, Bfrag);

  size_t smem = (size_t)2 * ROWS * LDK * 2 +
                (512 + 512 + ROWS + ROWS + TOKS * 64 + 256 + 256) * sizeof(float);
  int nblocks = (8 * 2048 * 8) / ROWS;   // 131072 rows / 64 = 2048
  expert_graph_conv_kernel<<<nblocks, 256, smem, stream>>>(
      x, bn, bs, wm, bm, al, Bfrag, out);
}
